// LSTMModel_39771397161368
// MI455X (gfx1250) — matrix operand order repaired and checked by reading, not compiled, not run on hardware
//
#include <hip/hip_runtime.h>

#define T_STEPS 512
#define IN_DIM  64
#define HID     128
#define LDSS    136   // padded row stride in f16 (16B-aligned chunks, conflict-minimal)

typedef __attribute__((ext_vector_type(16))) _Float16 v16h;
typedef __attribute__((ext_vector_type(8)))  float    v8f;

union FragU { uint4 q[2]; v16h h; };

// B-matrix fragment (32x16 f16) from GLOBAL: lane L -> column n = L&15;
// chunk0: K = (L>>4)*8 .. +7 ; chunk1: same +16, in row-major W (n, k) -> two 16B loads.
__device__ __forceinline__ v16h load_bfrag(const _Float16* __restrict__ W, int ldk,
                                           int n0, int kbase, int lane) {
    int n = n0 + (lane & 15);
    int k = kbase + ((lane >> 4) << 3);
    const uint4* p = reinterpret_cast<const uint4*>(W + (size_t)n * ldk + k);
    FragU f; f.q[0] = p[0]; f.q[1] = p[2];
    return f.h;
}

// B-matrix fragment from LDS-cached weights (padded stride LDSS).
__device__ __forceinline__ v16h load_bfrag_l(const _Float16* Wl, int n0, int kbase, int lane) {
    int n = n0 + (lane & 15);
    int k = kbase + ((lane >> 4) << 3);
    const _Float16* p = Wl + n * LDSS + k;
    FragU f;
    f.q[0] = *reinterpret_cast<const uint4*>(p);
    f.q[1] = *reinterpret_cast<const uint4*>(p + 16);
    return f.h;
}

// A-matrix fragment (16x32 f16) from LDS h-state: lane L -> row m = L&15;
// chunk0: K = kbase + (L>>4)*8 .. +7 ; chunk1: same +16.
__device__ __forceinline__ v16h load_afrag_lds(const _Float16* Hs, int kbase, int lane) {
    int m   = lane & 15;
    int off = (lane >> 4) << 3;
    const _Float16* p = Hs + m * LDSS + kbase + off;
    FragU f;
    f.q[0] = *reinterpret_cast<const uint4*>(p);
    f.q[1] = *reinterpret_cast<const uint4*>(p + 16);
    return f.h;
}

// A-matrix fragment of x_t (f32 in global, converted on the fly to f16).
__device__ __forceinline__ v16h load_afrag_x(const float* __restrict__ xb, int t,
                                             int kbase, int lane) {
    int m   = lane & 15;
    int off = (lane >> 4) << 3;
    const float* p = xb + (size_t)m * (T_STEPS * IN_DIM) + (size_t)t * IN_DIM + kbase + off;
    float4 a = *reinterpret_cast<const float4*>(p);
    float4 b = *reinterpret_cast<const float4*>(p + 4);
    float4 c = *reinterpret_cast<const float4*>(p + 16);
    float4 d = *reinterpret_cast<const float4*>(p + 20);
    v16h r;
    r[0]  = (_Float16)a.x; r[1]  = (_Float16)a.y; r[2]  = (_Float16)a.z; r[3]  = (_Float16)a.w;
    r[4]  = (_Float16)b.x; r[5]  = (_Float16)b.y; r[6]  = (_Float16)b.z; r[7]  = (_Float16)b.w;
    r[8]  = (_Float16)c.x; r[9]  = (_Float16)c.y; r[10] = (_Float16)c.z; r[11] = (_Float16)c.w;
    r[12] = (_Float16)d.x; r[13] = (_Float16)d.y; r[14] = (_Float16)d.z; r[15] = (_Float16)d.w;
    return r;
}

__device__ __forceinline__ float sigm(float x) {
    return 1.0f / (1.0f + __expf(-x));
}
__device__ __forceinline__ float tanh_fast(float x) {
    x = fminf(fmaxf(x, -15.0f), 15.0f);
    float e = __expf(2.0f * x);
    return (e - 1.0f) / (e + 1.0f);
}

// One-shot f32 -> f16 weight conversion into workspace.
__global__ void __launch_bounds__(256) convert_weights_kernel(
    const float* __restrict__ Wih0, const float* __restrict__ Whh0,
    const float* __restrict__ Wih1, const float* __restrict__ Whh1,
    _Float16* __restrict__ ws) {
    int i = blockIdx.x * 256 + threadIdx.x;
    if (i < 32768)        ws[i] = (_Float16)Wih0[i];                 // 512x64
    else if (i < 98304)   ws[i] = (_Float16)Whh0[i - 32768];         // 512x128
    else if (i < 163840)  ws[i] = (_Float16)Wih1[i - 98304];         // 512x128
    else if (i < 229376)  ws[i] = (_Float16)Whh1[i - 163840];        // 512x128
}

__global__ void __launch_bounds__(256) lstm2_wmma_kernel(
    const float* __restrict__ x,
    const float* __restrict__ b0,  const float* __restrict__ b1,
    const float* __restrict__ Wout, const float* __restrict__ bout,
    const _Float16* __restrict__ Wih0h, const _Float16* __restrict__ Whh0h,
    const _Float16* __restrict__ Wih1h, const _Float16* __restrict__ Whh1h,
    float* __restrict__ out) {

    // h-state (static LDS) + LDS-cached weights (dynamic LDS, 272KB of the WGP's 320KB)
    __shared__ _Float16 h0s[16 * LDSS];
    __shared__ _Float16 h1s[16 * LDSS];
    extern __shared__ __align__(16) _Float16 wlds[];   // [2][512*LDSS]
    _Float16* Whh0L = wlds;                            // layer0 hidden weights
    _Float16* Wih1L = wlds + 512 * LDSS;               // layer1 input weights

    const int lane  = threadIdx.x & 31;
    const int wave  = threadIdx.x >> 5;          // 0..7, owns hidden cols [16w,16w+16)
    const int btile = blockIdx.x;                // 0..63, owns batch rows [16b,16b+16)
    const float* xb = x + (size_t)btile * 16 * T_STEPS * IN_DIM;

    for (int i = threadIdx.x; i < 16 * LDSS; i += 256) {
        h0s[i] = (_Float16)0.0f;
        h1s[i] = (_Float16)0.0f;
    }
    // Stage Whh0 and Wih1 into padded LDS (512 rows x 16 chunks of 8 f16 each).
    for (int c = threadIdx.x; c < 512 * 16; c += 256) {
        int row = c >> 4, col = (c & 15) << 3;
        *reinterpret_cast<uint4*>(Whh0L + row * LDSS + col) =
            *reinterpret_cast<const uint4*>(Whh0h + row * HID + col);
        *reinterpret_cast<uint4*>(Wih1L + row * LDSS + col) =
            *reinterpret_cast<const uint4*>(Wih1h + row * HID + col);
    }
    __syncthreads();

    // Per-lane bias values in C-fragment layout (same for all 8 C VGPRs).
    float bias0[4], bias1[4];
#pragma unroll
    for (int g = 0; g < 4; ++g) {
        int n = g * HID + wave * 16 + (lane & 15);
        bias0[g] = b0[n];
        bias1[g] = b1[n];
    }

    v8f c0 = {}; v8f c1 = {};          // cell states: stay in registers for all 512 steps
    v8f h0new = {}, h1new = {};
    const int wm = (lane >> 4) << 3;   // C-layout row offset (0 or 8)
    const int wn = wave * 16 + (lane & 15);

    // Opaque scalar offsets: the empty asm "modifies" them every iteration so the
    // loop-invariant weight B-fragment loads (global AND lds) cannot be hoisted
    // (hoisting = ~450 VGPRs of weights + scratch spills). Indexing keeps the
    // address-space inference intact: global stays GLOBAL, lds stays DS.
    int wofs = 0, lofs = 0;

    for (int t = 0; t < T_STEPS; ++t) {
        asm("" : "+s"(wofs), "+s"(lofs));
        const _Float16* W0 = Wih0h + wofs;     // layer0 input weights (global/L2)
        const _Float16* W3 = Whh1h + wofs;     // layer1 hidden weights (global/L2)
        const _Float16* L1 = Whh0L + lofs;     // layer0 hidden weights (LDS)
        const _Float16* L2 = Wih1L + lofs;     // layer1 input weights (LDS)

        // -------- layer 0: z0 = x_t @ Wih0^T + h0 @ Whh0^T + b0 --------
        v16h ax0 = load_afrag_x(xb, t, 0, lane);
        v16h ax1 = load_afrag_x(xb, t, 32, lane);
        v16h ah[4];
#pragma unroll
        for (int kb = 0; kb < 4; ++kb) ah[kb] = load_afrag_lds(h0s, kb * 32, lane);

        v8f acc[4];
#pragma unroll
        for (int g = 0; g < 4; ++g) {
            int N = g * HID + wave * 16;
            v8f c;
#pragma unroll
            for (int r = 0; r < 8; ++r) c[r] = bias0[g];
            c = __builtin_amdgcn_wmma_f32_16x16x32_f16(false, ax0, false,
                    load_bfrag(W0, IN_DIM, N, 0, lane),  (short)0, c, false, false);
            c = __builtin_amdgcn_wmma_f32_16x16x32_f16(false, ax1, false,
                    load_bfrag(W0, IN_DIM, N, 32, lane), (short)0, c, false, false);
#pragma unroll
            for (int kb = 0; kb < 4; ++kb)
                c = __builtin_amdgcn_wmma_f32_16x16x32_f16(false, ah[kb], false,
                        load_bfrag_l(L1, N, kb * 32, lane), (short)0, c, false, false);
            acc[g] = c;
        }
#pragma unroll
        for (int r = 0; r < 8; ++r) {           // gates in i,f,g,o order
            float ig = sigm(acc[0][r]);
            float fg = sigm(acc[1][r]);
            float gg = tanh_fast(acc[2][r]);
            float og = sigm(acc[3][r]);
            float cn = fg * c0[r] + ig * gg;
            c0[r]    = cn;
            h0new[r] = og * tanh_fast(cn);
        }
        __syncthreads();                         // all waves done reading old h0
#pragma unroll
        for (int r = 0; r < 8; ++r) h0s[(r + wm) * LDSS + wn] = (_Float16)h0new[r];
        __syncthreads();                         // new h0 visible

        // -------- layer 1: z1 = h0_new @ Wih1^T + h1 @ Whh1^T + b1 --------
        v16h ah0n[4], ah1[4];
#pragma unroll
        for (int kb = 0; kb < 4; ++kb) {
            ah0n[kb] = load_afrag_lds(h0s, kb * 32, lane);
            ah1[kb]  = load_afrag_lds(h1s, kb * 32, lane);
        }
#pragma unroll
        for (int g = 0; g < 4; ++g) {
            int N = g * HID + wave * 16;
            v8f c;
#pragma unroll
            for (int r = 0; r < 8; ++r) c[r] = bias1[g];
#pragma unroll
            for (int kb = 0; kb < 4; ++kb)
                c = __builtin_amdgcn_wmma_f32_16x16x32_f16(false, ah0n[kb], false,
                        load_bfrag_l(L2, N, kb * 32, lane), (short)0, c, false, false);
#pragma unroll
            for (int kb = 0; kb < 4; ++kb)
                c = __builtin_amdgcn_wmma_f32_16x16x32_f16(false, ah1[kb], false,
                        load_bfrag(W3, HID, N, kb * 32, lane), (short)0, c, false, false);
            acc[g] = c;
        }
#pragma unroll
        for (int r = 0; r < 8; ++r) {
            float ig = sigm(acc[0][r]);
            float fg = sigm(acc[1][r]);
            float gg = tanh_fast(acc[2][r]);
            float og = sigm(acc[3][r]);
            float cn = fg * c1[r] + ig * gg;
            c1[r]    = cn;
            h1new[r] = og * tanh_fast(cn);
        }
        __syncthreads();
#pragma unroll
        for (int r = 0; r < 8; ++r) h1s[(r + wm) * LDSS + wn] = (_Float16)h1new[r];
        __syncthreads();
    }

    // -------- output projection: out = h1 @ Wout^T + bout  (OUT_DIM = 1) --------
    if (threadIdx.x < 16) {
        int m = threadIdx.x;
        float s = bout[0];
#pragma unroll 8
        for (int n = 0; n < HID; ++n) s += (float)h1s[m * LDSS + n] * Wout[n];
        out[btile * 16 + m] = s;
    }
}

extern "C" void kernel_launch(void* const* d_in, const int* in_sizes, int n_in,
                              void* d_out, int out_size, void* d_ws, size_t ws_size,
                              hipStream_t stream) {
    const float* x    = (const float*)d_in[0];
    const float* Wih0 = (const float*)d_in[1];
    const float* Whh0 = (const float*)d_in[2];
    const float* b0   = (const float*)d_in[3];
    const float* Wih1 = (const float*)d_in[4];
    const float* Whh1 = (const float*)d_in[5];
    const float* b1   = (const float*)d_in[6];
    const float* Wout = (const float*)d_in[7];
    const float* bout = (const float*)d_in[8];
    float* out = (float*)d_out;

    _Float16* ws = (_Float16*)d_ws;
    _Float16* Wih0h = ws;
    _Float16* Whh0h = ws + 32768;
    _Float16* Wih1h = ws + 98304;
    _Float16* Whh1h = ws + 163840;

    // 229376 f16 weight elements total
    convert_weights_kernel<<<896, 256, 0, stream>>>(Wih0, Whh0, Wih1, Whh1, ws);

    // 64 batch tiles x 8 waves; each WG runs the full T=512 recurrence.
    // Dynamic LDS: 2 matrices x 512 rows x LDSS f16 = 278528 bytes (of 320KB/WGP).
    size_t ldsBytes = (size_t)2 * 512 * LDSS * sizeof(_Float16);
    lstm2_wmma_kernel<<<64, 256, ldsBytes, stream>>>(x, b0, b1, Wout, bout,
                                                     Wih0h, Whh0h, Wih1h, Whh1h, out);
}
